// LSTM_noforget2_9560597201287
// MI455X (gfx1250) — matrix operand order repaired and checked by reading, not compiled, not run on hardware
//
#include <hip/hip_runtime.h>
#include <hip/hip_bf16.h>

constexpr int BATCH = 64;
constexpr int TSEQ  = 2048;
constexpr int NIN   = 256;
constexpr int NHID  = 512;
constexpr int NOUT  = 256;
constexpr int KTOT  = NIN + NHID;   // 768
constexpr int NKT   = KTOT / 32;    // 24 K-tiles total (8 from x, 16 from h)
constexpr int NWG   = 16;           // persistent workgroups in recurrent kernel
constexpr int JPW   = NHID / NWG;   // 32 hidden units owned per workgroup

typedef __attribute__((ext_vector_type(16))) __bf16 bf16x16;
typedef __attribute__((ext_vector_type(8)))  float  f32x8;
typedef unsigned short u16;
typedef unsigned int   u32;

__device__ __forceinline__ u16 f32_to_bf16(float f) {
  u32 u = __builtin_bit_cast(u32, f);
  u32 r = 0x7FFFu + ((u >> 16) & 1u);   // round-to-nearest-even
  return (u16)((u + r) >> 16);
}

__device__ __forceinline__ f32x8 wmma_bf16(bf16x16 a, bf16x16 b, f32x8 c) {
  return __builtin_amdgcn_wmma_f32_16x16x32_bf16(false, a, false, b, (short)0, c,
                                                 false, false);
}

// All operands are pre-swizzled into per-lane fragment layout:
// fragment = [lane 0..31][16 bf16] = 32 contiguous bytes per lane.
// Loading one fragment = two b128 loads per lane, no packing ALU.
__device__ __forceinline__ bf16x16 load_frag(const u16* __restrict__ p) {
  union { uint4 q[2]; bf16x16 v; } f;
  f.q[0] = *(const uint4*)(p);
  f.q[1] = *(const uint4*)(p + 8);
  return f.v;
}

// A-fragment swizzle (16x32 bf16, rows M, cols K), wave32 WMMA layout:
//   lane = hs*16 + m, hs = (kl>>3)&1
//   pos  = (kl&7) + ((kl>>4)&1)*8        (halves within the lane's 32B)
// B-fragment swizzle (32x16 bf16, rows K, cols N):
//   lane = ((kl>>3)&1)*16 + n ; pos = (kl&7) + ((kl>>4)&1)*8

// ---------------------------------------------------------------------------
// Persistent recurrent kernel: 16 WGs x 128 threads (4 waves).
// WG wg owns hidden units [wg*32, wg*32+32) for all three gates; its weight
// slice (144 KB of pre-swizzled B fragments) lives in LDS for the whole scan.
// Wave w computes batch rows [16w, 16w+16). h is kept in A-fragment layout.
// ---------------------------------------------------------------------------
__global__ __launch_bounds__(128)
void lstm_recurrent(const u16* __restrict__ xfrag,   // [mtile][T][8kt][32][16]
                    const u16* __restrict__ wfrag,   // [g][kt][ntg(32)][32][16]
                    const float* __restrict__ b_i,
                    const float* __restrict__ b_g,
                    const float* __restrict__ b_o,
                    u16* __restrict__ hbuf,          // [2][mtile][16kt][32][16]
                    u32* __restrict__ syncp) {
  extern __shared__ u16 lds_w[];                     // [g][kt][nt(2)][32][16]
  const int wg   = blockIdx.x;
  const int tid  = threadIdx.x;
  const int lane = tid & 31;
  const int wave = tid >> 5;                         // M-tile (batch/16)
  const int j0   = wg * JPW;
  const int nloc = lane & 15;
  const int hs   = (lane >> 4) & 1;

  // Stage this WG's 144 weight fragments into LDS (contiguous u32 copies).
  {
    const u32* __restrict__ src32 = (const u32*)wfrag;
    u32* __restrict__ dst32 = (u32*)lds_w;
    for (int idx = tid; idx < 3 * NKT * 2 * 256; idx += blockDim.x) {
      const int dword = idx & 255;        // 256 dwords per fragment
      const int frag  = idx >> 8;         // (g*NKT + kt)*2 + nt
      const int nt    = frag & 1;
      const int gkt   = frag >> 1;
      dst32[(size_t)frag * 256 + dword] =
          src32[((size_t)gkt * 32 + wg * 2 + nt) * 256 + dword];
    }
  }
  __syncthreads();

  float bias[3][2];
#pragma unroll
  for (int nt = 0; nt < 2; ++nt) {
    bias[0][nt] = b_i[j0 + nt * 16 + nloc];
    bias[1][nt] = b_g[j0 + nt * 16 + nloc];
    bias[2][nt] = b_o[j0 + nt * 16 + nloc];
  }
  // h store swizzle constants (store in A-fragment layout for next step).
  const int hs_h  = (nloc >> 3) & 1;             // lane-half of the h fragment
  const int pos0  = nloc & 7;                    // + nt*8

  for (int t = 0; t < TSEQ; ++t) {
    const u16* hcur = hbuf + (size_t)(t & 1) * (BATCH * NHID);
    u16*       hnxt = hbuf + (size_t)((t & 1) ^ 1) * (BATCH * NHID);

    if (t + 1 < TSEQ) {  // prefetch next timestep's 8KB of x fragments
      __builtin_prefetch(
          xfrag + (((size_t)wave * TSEQ + (t + 1)) * 8) * 512 + lane * 128, 0, 1);
    }

    f32x8 acc[3][2];
    const f32x8 zero = {0.f, 0.f, 0.f, 0.f, 0.f, 0.f, 0.f, 0.f};
#pragma unroll
    for (int g = 0; g < 3; ++g)
#pragma unroll
      for (int nt = 0; nt < 2; ++nt) acc[g][nt] = zero;

    // K-tiles 0..7: x contribution (coalesced b128 fragment loads).
    const u16* xbase = xfrag + (((size_t)wave * TSEQ + t) * 8) * 512 + lane * 16;
#pragma unroll 2
    for (int kt = 0; kt < 8; ++kt) {
      const bf16x16 a = load_frag(xbase + kt * 512);
#pragma unroll
      for (int g = 0; g < 3; ++g) {
#pragma unroll
        for (int nt = 0; nt < 2; ++nt) {
          const bf16x16 bm =
              load_frag(lds_w + (((size_t)(g * NKT + kt) * 2 + nt) * 32 + lane) * 16);
          acc[g][nt] = wmma_bf16(a, bm, acc[g][nt]);
        }
      }
    }
    // K-tiles 8..23: h contribution (h stored in fragment layout).
    const u16* hbase = hcur + ((size_t)wave * 16) * 512 + lane * 16;
#pragma unroll 2
    for (int kt = 0; kt < 16; ++kt) {
      const bf16x16 a = load_frag(hbase + kt * 512);
#pragma unroll
      for (int g = 0; g < 3; ++g) {
#pragma unroll
        for (int nt = 0; nt < 2; ++nt) {
          const bf16x16 bm = load_frag(
              lds_w + (((size_t)(g * NKT + 8 + kt) * 2 + nt) * 32 + lane) * 16);
          acc[g][nt] = wmma_bf16(a, bm, acc[g][nt]);
        }
      }
    }

    // Fused gates in registers; write h_new in A-fragment layout.
    // b = wave*16 + v + hs*8 -> mtile=wave, m=v+hs*8
    // j = j0 + nt*16 + nloc  -> kt_h=wg, kl=nt*16+nloc
#pragma unroll
    for (int nt = 0; nt < 2; ++nt) {
#pragma unroll
      for (int v = 0; v < 8; ++v) {
        const float pi = acc[0][nt][v] + bias[0][nt];
        const float pg = acc[1][nt][v] + bias[1][nt];
        const float po = acc[2][nt][v] + bias[2][nt];
        const float it = 1.0f / (1.0f + __expf(-pi));
        const float eg = __expf(2.0f * fabsf(pg));
        const float gt = copysignf(1.0f - 2.0f / (eg + 1.0f), pg);
        const float ot = 1.0f / (1.0f + __expf(-po));
        const float ct = it * gt;
        const float ec = __expf(2.0f * fabsf(ct));
        const float hn = ot * copysignf(1.0f - 2.0f / (ec + 1.0f), ct);
        const int m = v + hs * 8;
        const size_t off =
            (((size_t)(wave * 16 + wg)) * 32 + hs_h * 16 + m) * 16 + pos0 + nt * 8;
        hnxt[off] = f32_to_bf16(hn);
      }
    }

    // Device-scope barrier across the 16 persistent workgroups.
    __threadfence();
    __syncthreads();
    if (tid == 0) {
      __hip_atomic_fetch_add(syncp, 1u, __ATOMIC_RELEASE, __HIP_MEMORY_SCOPE_AGENT);
    }
    if ((tid & 31) == 0) {  // one acquiring spinner per wave
      const u32 target = (u32)(NWG * (t + 1));
      while (__hip_atomic_load(syncp, __ATOMIC_ACQUIRE, __HIP_MEMORY_SCOPE_AGENT) < target) {
        __builtin_amdgcn_s_sleep(1);
      }
    }
    __syncthreads();
  }
}

// Readout: out = h_last @ W_out + b_out. One wave per 16x16 C tile.
__global__ __launch_bounds__(32)
void lstm_readout(const u16* __restrict__ hlast,    // fragment layout, buffer 0
                  const u16* __restrict__ woutfrag, // [kt(16)][ntg(16)][32][16]
                  const float* __restrict__ b_out,
                  float* __restrict__ out) {
  const int mt = blockIdx.x & 3;
  const int nt = blockIdx.x >> 2;
  const int lane = threadIdx.x & 31;
  const int nloc = lane & 15;
  const int hs   = (lane >> 4) & 1;

  f32x8 acc = {0.f, 0.f, 0.f, 0.f, 0.f, 0.f, 0.f, 0.f};
#pragma unroll 4
  for (int kt = 0; kt < NHID / 32; ++kt) {
    const bf16x16 a = load_frag(hlast + (((size_t)mt * 16 + kt) * 32 + lane) * 16);
    const bf16x16 b =
        load_frag(woutfrag + (((size_t)kt * 16 + nt) * 32 + lane) * 16);
    acc = wmma_bf16(a, b, acc);
  }
  const float bv = b_out[nt * 16 + nloc];
#pragma unroll
  for (int v = 0; v < 8; ++v) {
    const int brow = mt * 16 + v + hs * 8;
    out[(size_t)brow * NOUT + nt * 16 + nloc] = acc[v] + bv;
  }
}

// x (f32 [B][T][NIN]) -> bf16 A-fragment layout [mtile][t][kt][lane][16].
__global__ void lstm_convert_x(const float* __restrict__ x,
                               u16* __restrict__ xfrag) {
  long i = (long)blockIdx.x * blockDim.x + threadIdx.x;
  const long stride = (long)gridDim.x * blockDim.x;
  const long n = (long)BATCH * TSEQ * NIN;
  for (; i < n; i += stride) {
    const int c  = (int)(i % NIN);
    const long bt = i / NIN;
    const int t  = (int)(bt % TSEQ);
    const int b  = (int)(bt / TSEQ);
    const int kt = c >> 5, kl = c & 31;
    const int hsf  = (kl >> 3) & 1;
    const int pos  = (kl & 7) + ((kl >> 4) & 1) * 8;
    const int lane = hsf * 16 + (b & 15);
    const size_t off =
        ((((size_t)(b >> 4) * TSEQ + t) * 8 + kt) * 32 + lane) * 16 + pos;
    xfrag[off] = f32_to_bf16(x[i]);
  }
}

// Gate weights (f32 [768][512] each) -> bf16 B-fragment layout
// [g][kt(24)][ntg(32)][lane(32)][pos(16)].
__global__ void lstm_convert_gates(const float* __restrict__ wi,
                                   const float* __restrict__ wgt,
                                   const float* __restrict__ wo,
                                   u16* __restrict__ wfrag) {
  const int idx = blockIdx.x * blockDim.x + threadIdx.x;
  if (idx >= KTOT * NHID) return;
  const int k = idx / NHID, j = idx % NHID;
  const int kt = k >> 5, kl = k & 31;
  const int lane = ((kl >> 3) & 1) * 16 + (j & 15);
  const int pos  = (kl & 7) + ((kl >> 4) & 1) * 8;
  const int ntg  = j >> 4;
  const float vals[3] = {wi[idx], wgt[idx], wo[idx]};
#pragma unroll
  for (int g = 0; g < 3; ++g) {
    wfrag[(((size_t)(g * NKT + kt) * 32 + ntg) * 32 + lane) * 16 + pos] =
        f32_to_bf16(vals[g]);
  }
}

// W_out (f32 [512][256]) -> bf16 B-fragment layout [kt(16)][ntg(16)][32][16].
__global__ void lstm_convert_wout(const float* __restrict__ w,
                                  u16* __restrict__ wfrag) {
  const int idx = blockIdx.x * blockDim.x + threadIdx.x;
  if (idx >= NHID * NOUT) return;
  const int k = idx / NOUT, j = idx % NOUT;
  const int kt = k >> 5, kl = k & 31;
  const int lane = ((kl >> 3) & 1) * 16 + (j & 15);
  const int pos  = (kl & 7) + ((kl >> 4) & 1) * 8;
  const int ntg  = j >> 4;
  wfrag[(((size_t)kt * 16 + ntg) * 32 + lane) * 16 + pos] = f32_to_bf16(w[idx]);
}

// Zero h0 (buffer 0), reset barrier counter, pass through `hidden` output.
__global__ void lstm_init(u16* __restrict__ hbuf, u32* __restrict__ syncp,
                          const float* __restrict__ hidden,
                          float* __restrict__ out_hidden) {
  const int i = blockIdx.x * blockDim.x + threadIdx.x;
  if (i < BATCH * NHID) hbuf[i] = 0;
  if (i < NHID) out_hidden[i] = hidden[i];
  if (i == 0) *syncp = 0u;
}

extern "C" void kernel_launch(void* const* d_in, const int* in_sizes, int n_in,
                              void* d_out, int out_size, void* d_ws, size_t ws_size,
                              hipStream_t stream) {
  const float* x      = (const float*)d_in[0];
  const float* hidden = (const float*)d_in[1];
  const float* W_i    = (const float*)d_in[2];
  const float* b_i    = (const float*)d_in[3];
  const float* W_g    = (const float*)d_in[4];
  const float* b_g    = (const float*)d_in[5];
  const float* W_o    = (const float*)d_in[6];
  const float* b_o    = (const float*)d_in[7];
  const float* W_out  = (const float*)d_in[8];
  const float* b_out  = (const float*)d_in[9];
  float* out = (float*)d_out;

  char* ws = (char*)d_ws;
  const size_t xf_bytes   = (size_t)BATCH * TSEQ * NIN * 2;  // 64 MiB
  const size_t wf_bytes   = (size_t)3 * KTOT * NHID * 2;     // 2.25 MiB
  const size_t wout_bytes = (size_t)NHID * NOUT * 2;         // 256 KiB
  const size_t hbuf_bytes = 2ull * BATCH * NHID * 2;         // 128 KiB
  u16* xfrag  = (u16*)ws;
  u16* wfrag  = (u16*)(ws + xf_bytes);
  u16* woutf  = (u16*)(ws + xf_bytes + wf_bytes);
  u16* hbuf   = (u16*)(ws + xf_bytes + wf_bytes + wout_bytes);
  u32* syncp  = (u32*)(ws + xf_bytes + wf_bytes + wout_bytes + hbuf_bytes);

  lstm_convert_x<<<2048, 256, 0, stream>>>(x, xfrag);
  lstm_convert_gates<<<(KTOT * NHID + 255) / 256, 256, 0, stream>>>(W_i, W_g, W_o,
                                                                    wfrag);
  lstm_convert_wout<<<(NHID * NOUT + 255) / 256, 256, 0, stream>>>(W_out, woutf);
  lstm_init<<<(BATCH * NHID + 255) / 256, 256, 0, stream>>>(hbuf, syncp, hidden,
                                                            out + BATCH * NOUT);
  lstm_recurrent<<<NWG, 128, (size_t)(3 * KTOT * JPW * 2), stream>>>(
      xfrag, wfrag, b_i, b_g, b_o, hbuf, syncp);
  lstm_readout<<<64, 32, 0, stream>>>(hbuf, woutf, b_out, out);
}
